// MoEBlock_25082609009227
// MI455X (gfx1250) — matrix operand order repaired and checked by reading, not compiled, not run on hardware
//
#include <hip/hip_runtime.h>
#include <hip/hip_bf16.h>
#include <math.h>

// ---------------- problem constants (from reference) ----------------
#define NTOK   8192      // B*T = 4*2048
#define DDIM   1024
#define HDIM   4096
#define NEXP   8
#define BM     64        // tokens per expert tile (4 M-halves of 16)
#define HC     64        // h-chunk width through LDS
#define TILES_MAX (NTOK / BM)   // 128 tiles per expert worst case

typedef __attribute__((ext_vector_type(16))) __bf16  v16bf;
typedef __attribute__((ext_vector_type(8)))  float   v8f;
typedef __attribute__((ext_vector_type(4)))  unsigned int uint4e;
typedef __attribute__((ext_vector_type(4)))  float   float4e;
typedef __attribute__((ext_vector_type(4)))  unsigned int u32x4;
typedef __attribute__((ext_vector_type(8)))  int     i32x8;
typedef __attribute__((ext_vector_type(4)))  int     i32x4;

// ---------------- helpers ----------------
__device__ __forceinline__ unsigned short f32_to_bf16(float f) {
  unsigned int u = __float_as_uint(f);
  u = (u + 0x7FFFu + ((u >> 16) & 1u)) >> 16;       // round-to-nearest-even
  return (unsigned short)u;
}

__device__ __forceinline__ unsigned int bf16pack2(float lo, float hi) {
  unsigned int ul = __float_as_uint(lo), uh = __float_as_uint(hi);
  ul = (ul + 0x7FFFu + ((ul >> 16) & 1u)) >> 16;
  uh = (uh + 0x7FFFu + ((uh >> 16) & 1u)) >> 16;
  return (uh << 16) | (ul & 0xFFFFu);
}

__device__ __forceinline__ float gelu_exact(float v) {
  return 0.5f * v * (1.0f + erff(v * 0.70710678118654752f));
}

__device__ __forceinline__ v8f wmma_bf16(v16bf a, v16bf b, v8f c) {
  return __builtin_amdgcn_wmma_f32_16x16x32_bf16(false, a, false, b, (short)0, c,
                                                 false, false);
}

// A operand (16x32 bf16) from LDS, contiguous k order (k = 16*(lane>>4) + s) -> 2x ds_load_b128
__device__ __forceinline__ v16bf load_A_lds(const unsigned short* base, int lda,
                                            int m0, int k0, int lane) {
  int m  = m0 + (lane & 15);
  int kb = (lane >> 4) << 4;                 // 0 or 16
  const unsigned short* p = base + m * lda + k0 + kb;
  union { uint4e q[2]; v16bf v; } u;
  u.q[0] = *(const uint4e*)(p);              // K = kb .. kb+7
  u.q[1] = *(const uint4e*)(p + 8);          // K = kb+8 .. kb+15
  return u.v;
}

// B operand (32x16 bf16) from the pre-packed tile buffer: each tile is 512
// bf16 (1 KB) stored in exact per-lane operand order -> 2x global_load_b128.
__device__ __forceinline__ v16bf load_B_packed(const unsigned short* __restrict__ Wp,
                                               size_t tile, int lane) {
  const unsigned short* p = Wp + tile * 512 + (size_t)lane * 16;
  union { uint4e q[2]; v16bf v; } u;
  u.q[0] = ((const uint4e*)p)[0];
  u.q[1] = ((const uint4e*)p)[1];
  return u.v;
}

// ---------------- kernel 0: zero out + counters ----------------
__global__ void init_kernel(float* __restrict__ out, int out_size,
                            int* __restrict__ cnt) {
  int idx = blockIdx.x * blockDim.x + threadIdx.x;
  if (idx < out_size) out[idx] = 0.0f;
  if (idx < NEXP) cnt[idx] = 0;
}

// ---------------- kernel 0b: x fp32 -> bf16 (coalesced stream) ----------------
__global__ void convert_x_kernel(const float* __restrict__ x,
                                 unsigned short* __restrict__ xb) {
  size_t idx = ((size_t)blockIdx.x * blockDim.x + threadIdx.x) * 4;
  if (idx < (size_t)NTOK * DDIM) {
    float4e v = *(const float4e*)(x + idx);
    *(unsigned int*)(xb + idx)     = bf16pack2(v.x, v.y);
    *(unsigned int*)(xb + idx + 2) = bf16pack2(v.z, v.w);
  }
}

// ---------------- kernel 0c: pack (E,K,N) fp32 weights into bf16 B-tiles ----
// Tile order within an expert: tile = ntile*(K/32) + kt. Within a tile, dword
// (lane,v) holds W[k0+kb+2v +{0,1}, n0+(lane&15)], kb=(lane>>4)*16 — the WMMA
// B operand register image.
__global__ void pack_w_kernel(const float* __restrict__ W,
                              unsigned short* __restrict__ Wp,
                              int Kdim, int Ndim) {
  size_t total = (size_t)NEXP * Kdim * Ndim / 2;     // output dwords
  size_t nkt = (size_t)(Kdim >> 5);
  size_t ntl = (size_t)(Ndim >> 4);
  unsigned int* out = (unsigned int*)Wp;
  for (size_t idx = (size_t)blockIdx.x * blockDim.x + threadIdx.x; idx < total;
       idx += (size_t)gridDim.x * blockDim.x) {
    int    lane = (int)(idx & 31);                   // fastest -> coalesced reads
    int    v    = (int)((idx >> 5) & 7);
    size_t tile = idx >> 8;                          // global tile id
    size_t e    = tile / (nkt * ntl);
    size_t tl   = tile % (nkt * ntl);
    size_t ntile = tl / nkt;
    size_t kt    = tl % nkt;
    int n  = (int)(ntile << 4) + (lane & 15);
    int kb = (lane >> 4) << 4;
    int k  = (int)(kt << 5) + kb + 2 * v;
    const float* We = W + e * (size_t)Kdim * Ndim;
    float f0 = We[(size_t)k * Ndim + n];
    float f1 = We[(size_t)(k + 1) * Ndim + n];
    out[tile * 256 + (size_t)lane * 8 + v] = bf16pack2(f0, f1);
  }
}

// ---------------- kernel 1: router (one wave per token) ----------------
__global__ __launch_bounds__(256)
void router_kernel(const float* __restrict__ x, const float* __restrict__ Wr,
                   float* __restrict__ probs, int* __restrict__ cnt,
                   int* __restrict__ tok_list, float* __restrict__ gate_list) {
  __shared__ float wrs[DDIM * NEXP];   // 32 KB router weights
  int tid = threadIdx.x;
  for (int i = tid; i < DDIM * NEXP; i += 256) wrs[i] = Wr[i];
  __syncthreads();

  int lane = tid & 31;
  int w    = tid >> 5;
  int t    = blockIdx.x * 8 + w;
  const float* xr = x + (size_t)t * DDIM;

  float lg[NEXP];
#pragma unroll
  for (int e = 0; e < NEXP; ++e) lg[e] = 0.0f;

  for (int d = lane; d < DDIM; d += 32) {
    float xv = xr[d];
#pragma unroll
    for (int e = 0; e < NEXP; ++e) lg[e] += xv * wrs[d * NEXP + e];
  }
#pragma unroll
  for (int e = 0; e < NEXP; ++e) {
#pragma unroll
    for (int off = 16; off > 0; off >>= 1)
      lg[e] += __shfl_xor(lg[e], off, 32);
  }

  if (lane == 0) {
    float mx = lg[0];
#pragma unroll
    for (int e = 1; e < NEXP; ++e) mx = fmaxf(mx, lg[e]);
    float p[NEXP], s = 0.0f;
#pragma unroll
    for (int e = 0; e < NEXP; ++e) { p[e] = __expf(lg[e] - mx); s += p[e]; }
    float inv = 1.0f / s;
#pragma unroll
    for (int e = 0; e < NEXP; ++e) { p[e] *= inv; probs[(size_t)t * NEXP + e] = p[e]; }

    int i1 = 0;
#pragma unroll
    for (int e = 1; e < NEXP; ++e) if (p[e] > p[i1]) i1 = e;
    int i2 = (i1 == 0) ? 1 : 0;
#pragma unroll
    for (int e = 0; e < NEXP; ++e) if (e != i1 && p[e] > p[i2]) i2 = e;

    float denom = p[i1] + p[i2] + 1e-9f;
    float g1 = p[i1] / denom, g2 = p[i2] / denom;

    int pos1 = atomicAdd(&cnt[i1], 1);
    tok_list[i1 * NTOK + pos1]  = t;
    gate_list[i1 * NTOK + pos1] = g1;
    int pos2 = atomicAdd(&cnt[i2], 1);
    tok_list[i2 * NTOK + pos2]  = t;
    gate_list[i2 * NTOK + pos2] = g2;
  }
}

// ---------------- kernel 2: pad expert lists to multiple of BM ----------------
__global__ void pad_kernel(const int* __restrict__ cnt, int* __restrict__ tok_list,
                           float* __restrict__ gate_list) {
  int tid = threadIdx.x;
  int e = tid >> 6, j = tid & 63;            // 512 threads: 8 experts x 64 slots
  if (e >= NEXP) return;
  int c  = cnt[e];
  int ce = (c + BM - 1) & ~(BM - 1);
  int pos = c + j;
  if (pos < ce) {
    tok_list[e * NTOK + pos]  = -1;
    gate_list[e * NTOK + pos] = 0.0f;
  }
}

// ---------------- kernel 3: deterministic importance reduction ----------------
__global__ void imp_reduce_kernel(const float* __restrict__ probs,
                                  float* __restrict__ imp) {
  __shared__ float red[256];
  int e = blockIdx.x, tid = threadIdx.x;
  float s = 0.0f;
  for (int t = tid; t < NTOK; t += 256) s += probs[(size_t)t * NEXP + e];
  red[tid] = s;
  __syncthreads();
  for (int st = 128; st > 0; st >>= 1) {
    if (tid < st) red[tid] += red[tid + st];
    __syncthreads();
  }
  if (tid == 0) imp[e] = red[0];
}

// ---------------- kernel 4: load-balance loss scalar ----------------
__global__ void lb_final_kernel(const float* __restrict__ imp,
                                const int* __restrict__ cnt,
                                float* __restrict__ out_lb) {
  float si = 0.0f, sl = 0.0f;
  float im[NEXP], ld[NEXP];
#pragma unroll
  for (int e = 0; e < NEXP; ++e) {
    im[e] = imp[e]; ld[e] = (float)cnt[e];
    si += im[e]; sl += ld[e];
  }
  float lb = 0.0f;
#pragma unroll
  for (int e = 0; e < NEXP; ++e)
    lb += (im[e] / (si + 1e-9f)) * (ld[e] / (sl + 1e-9f));
  *out_lb = lb * (float)NEXP;
}

// ---------------- kernel 5: fused grouped expert MLP (WMMA bf16) ----------------
// 1 block per (expert, 64-token tile); 8 wave32 waves.
// Per wave: GEMM1 -> 2 h tiles sharing one A; GEMM2 -> 4x4 acc tiles (128 VGPRs)
// over a private 64-wide D slice, B tiles reused across the 4 M-halves.
__global__ __launch_bounds__(256)
void moe_expert_kernel(const unsigned short* __restrict__ xb,   // bf16 x
                       const unsigned short* __restrict__ W1p,  // packed bf16
                       const unsigned short* __restrict__ W2p,  // packed bf16
                       const int* __restrict__ tok_list,
                       const float* __restrict__ gate_list,
                       const int* __restrict__ cnt, float* __restrict__ out) {
  __shared__ unsigned short xa[BM * DDIM];   // 128 KB x-tile, bf16
  __shared__ unsigned short hb[BM * HC];     // 8 KB h-chunk, bf16
  __shared__ int   toks[BM];
  __shared__ float gates[BM];

  int e  = blockIdx.x / TILES_MAX;
  int tt = blockIdx.x % TILES_MAX;
  int c  = cnt[e];
  int ce = (c + BM - 1) & ~(BM - 1);
  if (tt * BM >= ce) return;                 // uniform across block

  int tid  = threadIdx.x;
  int lane = tid & 31;
  int w    = tid >> 5;

  if (tid < BM) {
    toks[tid]  = tok_list[e * NTOK + tt * BM + tid];
    gates[tid] = gate_list[e * NTOK + tt * BM + tid];
  }
  __syncthreads();

#if __has_builtin(__builtin_amdgcn_tensor_load_to_lds) && \
    __has_builtin(__builtin_amdgcn_s_wait_tensorcnt)
  // ---- TDM gather: 4 descriptors x 16 rows of 1024 bf16 each -> xa ----
  if (w == 0) {
    unsigned lds_base = (unsigned)(unsigned long long)(void*)xa;
    unsigned long long ga = (unsigned long long)(const void*)xb;
#pragma unroll
    for (int half = 0; half < 4; ++half) {
      unsigned id[8];
#pragma unroll
      for (int i = 0; i < 8; ++i) {
        int t0 = toks[half * 16 + 2 * i + 0];
        int t1 = toks[half * 16 + 2 * i + 1];
        unsigned r0 = (t0 < 0) ? 0u : (unsigned)t0;   // sentinel -> row 0 (masked later)
        unsigned r1 = (t1 < 0) ? 0u : (unsigned)t1;
        id[i] = (unsigned)__builtin_amdgcn_readfirstlane((int)(r0 | (r1 << 16)));
      }
      // group0: count=1, gather_mode=1 (bit31), 16-bit indices (bit30=0)
      u32x4 g0 = { 0x80000001u,
                   lds_base + (unsigned)(half * 16 * DDIM * 2),
                   (unsigned)(ga & 0xFFFFFFFFull),
                   (unsigned)((ga >> 32) & 0x01FFFFFFull) | 0x80000000u }; // type=2
      // group1: data_size=2B(code1); tensor_dim0=1024; tensor_dim1=NTOK;
      //         tile_dim0=1024; tile_dim1=16 valid rows; dim0_stride=1024
      i32x8 g1 = { (int)(1u << 16),
                   (int)(DDIM << 16),
                   (int)(NTOK << 16),
                   (int)(DDIM << 16),
                   16,
                   DDIM,
                   0, 0 };
      i32x4 g2 = { (int)id[0], (int)id[1], (int)id[2], (int)id[3] };
      i32x4 g3 = { (int)id[4], (int)id[5], (int)id[6], (int)id[7] };
      i32x8 g4 = { 0, 0, 0, 0, 0, 0, 0, 0 };          // unused VADDR4 slot
      __builtin_amdgcn_tensor_load_to_lds(g0, g1, g2, g3, g4, 0);
    }
    __builtin_amdgcn_s_wait_tensorcnt(0);
  }
  __syncthreads();
#else
  // ---- fallback: plain gather copy of bf16 rows ----
  for (int i = tid * 8; i < BM * DDIM; i += 256 * 8) {
    int m = i >> 10;
    int d = i & (DDIM - 1);
    int tk = toks[m];
    uint4e val = {0u, 0u, 0u, 0u};
    if (tk >= 0) val = *(const uint4e*)(xb + (size_t)tk * DDIM + d);
    *(uint4e*)(xa + i) = val;
  }
  __syncthreads();
#endif

  const unsigned short* W1e = W1p + (size_t)e * DDIM * HDIM;  // packed tiles
  const unsigned short* W2e = W2p + (size_t)e * HDIM * DDIM;

  v8f acc[4][4];                             // 64 tokens x 64 D-cols per wave
#pragma unroll
  for (int mh = 0; mh < 4; ++mh)
#pragma unroll
    for (int nt = 0; nt < 4; ++nt) acc[mh][nt] = (v8f){0,0,0,0,0,0,0,0};

  int mh1  = w >> 1;                         // gemm1: M-half 0..3
  int nh1a = (w & 1) * 2;                    // gemm1: h-column tiles nh1a, nh1a+1

  for (int hc = 0; hc < HDIM; hc += HC) {
    // ---- GEMM1: h[64,HC] = gelu(x[64,1024] @ W1[:,hc:hc+HC]) ----
    v8f h0 = (v8f){0,0,0,0,0,0,0,0};
    v8f h1 = (v8f){0,0,0,0,0,0,0,0};
    size_t tb0 = (size_t)((hc >> 4) + nh1a)     * (DDIM >> 5);
    size_t tb1 = (size_t)((hc >> 4) + nh1a + 1) * (DDIM >> 5);
#pragma unroll 2
    for (int kk = 0; kk < DDIM; kk += 32) {
      __builtin_prefetch(W1e + (tb0 + (kk >> 5) + 2) * 512, 0, 3);
      v16bf A  = load_A_lds(xa, DDIM, mh1 * 16, kk, lane);
      v16bf B0 = load_B_packed(W1e, tb0 + (kk >> 5), lane);
      v16bf B1 = load_B_packed(W1e, tb1 + (kk >> 5), lane);
      h0 = wmma_bf16(A, B0, h0);             // A reused across nh pair
      h1 = wmma_bf16(A, B1, h1);
    }
    {
      int n  = lane & 15;
      int mo = (lane >> 4) << 3;
#pragma unroll
      for (int v = 0; v < 8; ++v) {
        int row = (mh1 * 16 + mo + v) * HC + n;
        hb[row + nh1a * 16]       = f32_to_bf16(gelu_exact(h0[v]));
        hb[row + (nh1a + 1) * 16] = f32_to_bf16(gelu_exact(h1[v]));
      }
    }
    __syncthreads();

    // ---- GEMM2: out[64, w*64 .. +64) += h[64,HC] @ W2[hc:hc+HC, :] ----
#pragma unroll
    for (int kk = 0; kk < HC; kk += 32) {
      size_t kt2 = (size_t)((hc + kk) >> 5);
      v16bf B[4];
#pragma unroll
      for (int nt = 0; nt < 4; ++nt) {
        size_t tile2 = (size_t)(w * 4 + nt) * (HDIM >> 5) + kt2;
        __builtin_prefetch(W2e + (tile2 + 2) * 512, 0, 3);
        B[nt] = load_B_packed(W2e, tile2, lane);
      }
#pragma unroll
      for (int mh = 0; mh < 4; ++mh) {
        v16bf A = load_A_lds(hb, HC, mh * 16, kk, lane);
#pragma unroll
        for (int nt = 0; nt < 4; ++nt)
          acc[mh][nt] = wmma_bf16(A, B[nt], acc[mh][nt]);  // B reused across mh
      }
    }
    __syncthreads();
  }

  // ---- epilogue: scale by gate, atomic-accumulate into out ----
  int n  = lane & 15;
  int mo = (lane >> 4) << 3;
#pragma unroll
  for (int mh = 0; mh < 4; ++mh) {
#pragma unroll
    for (int nt = 0; nt < 4; ++nt) {
#pragma unroll
      for (int v = 0; v < 8; ++v) {
        int m  = mh * 16 + mo + v;
        int tk = toks[m];
        if (tk >= 0) {
          float val = gates[m] * acc[mh][nt][v];
          atomicAdd(out + (size_t)tk * DDIM + w * 64 + nt * 16 + n, val);
        }
      }
    }
  }
}

// ---------------- host launcher ----------------
extern "C" void kernel_launch(void* const* d_in, const int* in_sizes, int n_in,
                              void* d_out, int out_size, void* d_ws, size_t ws_size,
                              hipStream_t stream) {
  const float* x  = (const float*)d_in[0];   // (B,T,D)
  const float* Wr = (const float*)d_in[1];   // (D,E)
  const float* W1 = (const float*)d_in[2];   // (E,D,H)
  const float* W2 = (const float*)d_in[3];   // (E,H,D)
  float* out = (float*)d_out;

  // ---- workspace carve-up (~146 MB total) ----
  char* ws = (char*)d_ws;
  int*   cnt       = (int*)ws;                                   // 8 ints
  float* imp       = (float*)(ws + 64);
  float* probs     = (float*)(ws + 128);                         // NTOK*NEXP f32
  int*   tok_list  = (int*)(ws + 128 + (size_t)NTOK * NEXP * 4);
  float* gate_list = (float*)((char*)tok_list + (size_t)NEXP * NTOK * 4);
  size_t off = (size_t)1 << 20;                                  // 1 MB control region
  unsigned short* xb  = (unsigned short*)(ws + off);
  off += (size_t)NTOK * DDIM * 2;                                // 16 MB
  unsigned short* W1p = (unsigned short*)(ws + off);
  off += (size_t)NEXP * DDIM * HDIM * 2;                         // 64 MB
  unsigned short* W2p = (unsigned short*)(ws + off);             // 64 MB

  int zb = (out_size + 255) / 256;
  init_kernel<<<zb, 256, 0, stream>>>(out, out_size, cnt);
  convert_x_kernel<<<(NTOK * DDIM / 4 + 255) / 256, 256, 0, stream>>>(x, xb);
  pack_w_kernel<<<16384, 256, 0, stream>>>(W1, W1p, DDIM, HDIM);
  pack_w_kernel<<<16384, 256, 0, stream>>>(W2, W2p, HDIM, DDIM);
  router_kernel<<<NTOK / 8, 256, 0, stream>>>(x, Wr, probs, cnt, tok_list, gate_list);
  pad_kernel<<<1, 512, 0, stream>>>(cnt, tok_list, gate_list);
  imp_reduce_kernel<<<NEXP, 256, 0, stream>>>(probs, imp);
  lb_final_kernel<<<1, 1, 0, stream>>>(imp, cnt, out + (out_size - 1));
  moe_expert_kernel<<<NEXP * TILES_MAX, 256, 0, stream>>>(xb, W1p, W2p, tok_list,
                                                          gate_list, cnt, out);
}
